// GlGraphConv_32220844655405
// MI455X (gfx1250) — matrix operand order repaired and checked by reading, not compiled, not run on hardware
//
#include <hip/hip_runtime.h>
#include <stdint.h>

// ---------------------------------------------------------------------------
// GlGraphConv for MI455X (gfx1250): fused 3x GEMM (bf16 WMMA, fp32 accum)
// + ring-adjacency attention mix + joint maxpool mix + bias.
// ---------------------------------------------------------------------------

typedef __attribute__((ext_vector_type(16))) __bf16     v16bf;
typedef __attribute__((ext_vector_type(8)))  float      v8f;
typedef __attribute__((ext_vector_type(4)))  unsigned   u32x4;

#define GC_N     16
#define GC_B     8192
#define GC_FIN   256
#define GC_FOUT  256
#define GC_BT    4            // batches per workgroup (W-fragment reuse factor)
#define GC_NEG   (-9e15f)

// Wp: packed bf16 W in WMMA B-operand layout.
// dword index D = (((m*16 + gt)*8 + ks)*32 + lane)*8 + j
// contents: lo = W[m][ks*32 + 2j + 16*(lane>=16)][gt*16 + lane%16]
//           hi = same with k+1
#define WP_DWORDS (3*16*8*32*8)   // 98304 dwords = 384 KB

__device__ __forceinline__ unsigned pack_bf16(float a, float b) {
  unsigned ua = __float_as_uint(a), ub = __float_as_uint(b);
  ua = (ua + 0x7FFFu + ((ua >> 16) & 1u)) >> 16;   // round-to-nearest-even
  ub = (ub + 0x7FFFu + ((ub >> 16) & 1u)) >> 16;
  return (ua & 0xFFFFu) | (ub << 16);
}

__device__ __forceinline__ v16bf make_v16bf(u32x4 lo, u32x4 hi) {
  union { u32x4 u[2]; v16bf v; } un;
  un.u[0] = lo; un.u[1] = hi;
  return un.v;
}

// cross-half (lane ^ 16) exchange via DS swizzle SWAPX16: xor=0x10,and=0x1f
__device__ __forceinline__ float swap16(float v) {
  return __int_as_float(__builtin_amdgcn_ds_swizzle(__float_as_int(v), 0x401f));
}

// ---------------------------------------------------------------------------
// Prep: A softmax, Q = 0.5*softmax(MM), and fp32->bf16 B-layout packing of W.
// ---------------------------------------------------------------------------
__global__ void gc_prep_kernel(const float* __restrict__ W,
                               const float* __restrict__ MM,
                               const float* __restrict__ e,
                               const int*   __restrict__ rows,
                               const int*   __restrict__ cols,
                               int nnz,
                               unsigned* __restrict__ Wp,
                               float* __restrict__ Aout,
                               float* __restrict__ Qout) {
  const int t = threadIdx.x;

  if (t < 16) {                       // adjacency attention row softmax
    float a[16];
    #pragma unroll
    for (int j = 0; j < 16; ++j) a[j] = GC_NEG;
    for (int s = 0; s < nnz; ++s)
      if (rows[s] == t) a[cols[s]] = e[s];
    float mx = a[0];
    #pragma unroll
    for (int j = 1; j < 16; ++j) mx = fmaxf(mx, a[j]);
    float sum = 0.f;
    #pragma unroll
    for (int j = 0; j < 16; ++j) { a[j] = __expf(a[j] - mx); sum += a[j]; }
    float inv = 1.f / sum;
    #pragma unroll
    for (int j = 0; j < 16; ++j)
      Aout[t * 16 + j] = (j == t) ? 0.f : a[j] * inv;   // zero self-attention
  } else if (t < 32) {                // Q = 0.5 * softmax(MM, axis=1)
    const int i = t - 16;
    float q[8];
    float mx = MM[i * 8];
    #pragma unroll
    for (int j = 1; j < 8; ++j) mx = fmaxf(mx, MM[i * 8 + j]);
    float sum = 0.f;
    #pragma unroll
    for (int j = 0; j < 8; ++j) { q[j] = __expf(MM[i * 8 + j] - mx); sum += q[j]; }
    float inv = 0.5f / sum;
    #pragma unroll
    for (int j = 0; j < 8; ++j) Qout[i * 8 + j] = q[j] * inv;
  }

  // pack W (3x256x256 fp32) into bf16 WMMA B-operand layout
  for (int d = t; d < WP_DWORDS; d += blockDim.x) {
    const int j    = d & 7;
    const int lane = (d >> 3) & 31;
    const int ks   = (d >> 8) & 7;
    const int gt   = (d >> 11) & 15;
    const int m    = d >> 15;
    const int k    = ks * 32 + 2 * j + ((lane >> 4) << 4);
    const int g    = gt * 16 + (lane & 15);
    const float w0 = W[(m * 256 + k)     * 256 + g];
    const float w1 = W[(m * 256 + k + 1) * 256 + g];
    Wp[d] = pack_bf16(w0, w1);
  }
}

// ---------------------------------------------------------------------------
// Main: 16 waves per WG; wave w owns gout tile [16w,16w+16); GC_BT batches.
// ---------------------------------------------------------------------------
__global__ __launch_bounds__(512, 1)
void gc_main_kernel(const float* __restrict__ x,
                    const unsigned* __restrict__ Wp,
                    const float* __restrict__ Aw,
                    const float* __restrict__ Qw,
                    const float* __restrict__ bias,
                    float* __restrict__ out) {
  __shared__ unsigned sX[GC_BT * GC_N * GC_FIN / 2];   // bf16 X tiles, 32 KB
  __shared__ float    sA[256];
  __shared__ float    sQ[128];

  const int t  = threadIdx.x;
  const int b0 = blockIdx.x * GC_BT;

  if (t < 256)      sA[t]       = Aw[t];
  else if (t < 384) sQ[t - 256] = Qw[t - 256];

  // cooperative fp32 -> bf16 conversion of the 4 X tiles into LDS
  {
    const float4* xb4 = (const float4*)(x + (size_t)b0 * GC_N * GC_FIN);
    u32x4* sX4 = (u32x4*)sX;
    #pragma unroll
    for (int it = 0; it < 4; ++it) {
      const int e8 = it * 4096 + t * 8;        // 8 consecutive fp32
      const float4 f0 = xb4[e8 / 4];
      const float4 f1 = xb4[e8 / 4 + 1];
      u32x4 p;
      p.x = pack_bf16(f0.x, f0.y); p.y = pack_bf16(f0.z, f0.w);
      p.z = pack_bf16(f1.x, f1.y); p.w = pack_bf16(f1.z, f1.w);
      sX4[it * 512 + t] = p;
    }
  }
  __syncthreads();

  const int lane = t & 31;
  const int w    = t >> 5;          // gout tile id 0..15
  const int half = lane >> 4;       // lane-half (C-layout row offset 8*half)
  const int g    = lane & 15;       // output column within tile

  v8f acc0[GC_BT], acc1[GC_BT], acc3[GC_BT];
  {
    const v8f z = {0.f, 0.f, 0.f, 0.f, 0.f, 0.f, 0.f, 0.f};
    #pragma unroll
    for (int bt = 0; bt < GC_BT; ++bt) { acc0[bt] = z; acc1[bt] = z; acc3[bt] = z; }
  }

  const u32x4* wp4 = (const u32x4*)Wp;
  const u32x4* sx4 = (const u32x4*)sX;
  const int aBase = (lane & 15) * 32 + 2 * half;  // uint4 units into sX
  const int wBase = w * 512 + lane * 2;       // uint4 units into Wp

  #pragma unroll
  for (int ks = 0; ks < 8; ++ks) {
    // B fragments for W0 / W1 / W2 (shared across the GC_BT batches)
    u32x4 bl[3], bh[3];
    #pragma unroll
    for (int m = 0; m < 3; ++m) {
      const int bi = m * 8192 + wBase + ks * 64;
      bl[m] = wp4[bi];
      bh[m] = wp4[bi + 1];
    }
    #pragma unroll
    for (int bt = 0; bt < GC_BT; ++bt) {
      const int ai = bt * 512 + aBase + ks * 4;
      const v16bf a = make_v16bf(sx4[ai], sx4[ai + 1]);
      acc0[bt] = __builtin_amdgcn_wmma_f32_16x16x32_bf16(
          false, a, false, make_v16bf(bl[0], bh[0]), (short)0, acc0[bt], false, false);
      acc1[bt] = __builtin_amdgcn_wmma_f32_16x16x32_bf16(
          false, a, false, make_v16bf(bl[1], bh[1]), (short)0, acc1[bt], false, false);
      acc3[bt] = __builtin_amdgcn_wmma_f32_16x16x32_bf16(
          false, a, false, make_v16bf(bl[2], bh[2]), (short)0, acc3[bt], false, false);
    }
  }

  // ---- epilogue: out = h0 + A*h1 + Q*maxpool2(h3) + bias -------------------
  const float bv = bias[w * 16 + g];
  float* outp = out + (size_t)b0 * GC_N * GC_FOUT;

  #pragma unroll
  for (int bt = 0; bt < GC_BT; ++bt) {
    // h1 rows 0..15 at this column (cross-half exchange for the other 8 rows)
    float rv[16];
    #pragma unroll
    for (int j = 0; j < 8; ++j) {
      const float own = acc1[bt][j];
      const float oth = swap16(own);
      rv[j]     = half ? oth : own;   // rows 0..7
      rv[j + 8] = half ? own : oth;   // rows 8..15
    }
    // maxpool over joint pairs: 8 pooled rows at this column
    float pl[8];
    #pragma unroll
    for (int k = 0; k < 4; ++k) {
      const float mo = fmaxf(acc3[bt][2 * k], acc3[bt][2 * k + 1]);
      const float ot = swap16(mo);
      pl[k]     = half ? ot : mo;     // pooled rows 0..3
      pl[k + 4] = half ? mo : ot;     // pooled rows 4..7
    }
    #pragma unroll
    for (int r = 0; r < 8; ++r) {
      const int i = r + 8 * half;     // output joint row
      float v = acc0[bt][r] + bv;
      #pragma unroll
      for (int j = 0; j < 16; ++j) v = fmaf(sA[i * 16 + j], rv[j], v);
      #pragma unroll
      for (int k = 0; k < 8; ++k)  v = fmaf(sQ[i * 8 + k], pl[k], v);
      outp[(size_t)(bt * GC_N + i) * GC_FOUT + w * 16 + g] = v;
    }
  }
}

// ---------------------------------------------------------------------------
extern "C" void kernel_launch(void* const* d_in, const int* in_sizes, int n_in,
                              void* d_out, int out_size, void* d_ws, size_t ws_size,
                              hipStream_t stream) {
  const float* x    = (const float*)d_in[0];   // [8192,16,256]
  const float* W    = (const float*)d_in[1];   // [3,256,256]
  const float* MM   = (const float*)d_in[2];   // [16,8]
  const float* e    = (const float*)d_in[3];   // [1,48]
  const float* bias = (const float*)d_in[4];   // [256]
  const int* rows   = (const int*)d_in[5];     // [48]
  const int* cols   = (const int*)d_in[6];     // [48]
  const int nnz = in_sizes[5];

  unsigned* Wp = (unsigned*)d_ws;                              // 384 KB
  float* Aw = (float*)((char*)d_ws + (size_t)WP_DWORDS * 4);   // 256 f32
  float* Qw = Aw + 256;                                        // 128 f32

  gc_prep_kernel<<<1, 256, 0, stream>>>(W, MM, e, rows, cols, nnz, Wp, Aw, Qw);
  gc_main_kernel<<<GC_B / GC_BT, 512, 0, stream>>>(
      x, Wp, Aw, Qw, bias, (float*)d_out);
}
